// MultiCellSSNModel_84378927497320
// MI455X (gfx1250) — matrix operand order repaired and checked by reading, not compiled, not run on hardware
//
#include <hip/hip_runtime.h>
#include <hip/hip_bf16.h>

#define N_TOT 8192      // N_CELL * B
#define L_TS  20
#define SUBSTEPS 2

typedef __attribute__((ext_vector_type(16))) __bf16 v16bf;
typedef __attribute__((ext_vector_type(8)))  float  v8f;

union TileU { v16bf v; float4 q[2]; };

// ---- one-time (per launch) fp32 -> bf16 conversion of W into d_ws ----
__global__ __launch_bounds__(256) void ssn_convert_w(const float* __restrict__ W,
                                                     __hip_bfloat16* __restrict__ Wb,
                                                     long long n) {
  long long i = (long long)blockIdx.x * blockDim.x + threadIdx.x;
  const long long stride = (long long)gridDim.x * blockDim.x;
  for (; i < n; i += stride) Wb[i] = __float2bfloat16(W[i]);
}

// ---- r = r0 ; out[0,:] = r0 ----
__global__ __launch_bounds__(256) void ssn_init(const float* __restrict__ r0,
                                                float* __restrict__ r,
                                                float* __restrict__ out) {
  const int i = blockIdx.x * 256 + threadIdx.x;
  const float v = r0[i];
  r[i] = v;
  out[i] = v;
}

// ---- one RK4 stage: x = r + c*ds*kprev ; kout = relu(W@x + h)^2 - x ----
// Workgroup = 8 waves, owns 16 output rows. Wave w covers K slice [w*1024, w*1024+1024)
// with 32 x v_wmma_f32_16x16x32_bf16; B columns all replicate the x segment, so the
// row-sums land in lanes 0/16 of the accumulator; cross-wave reduce via LDS.
__global__ __launch_bounds__(256) void ssn_stage(const __hip_bfloat16* __restrict__ Wb,
                                                 const float* __restrict__ r,
                                                 const float* __restrict__ kprev,
                                                 const float* __restrict__ h,
                                                 const float* __restrict__ t,
                                                 int interval, float c,
                                                 float* __restrict__ kout) {
  __shared__ __align__(16) __hip_bfloat16 xb[N_TOT];  // 16 KB stage input in bf16
  __shared__ float red[8][16];

  const int tid  = threadIdx.x;
  const int lane = tid & 31;
  const int wave = tid >> 5;

  const float ds = (t[interval + 1] - t[interval]) * (1.0f / SUBSTEPS);
  const float a  = c * ds;

  // Build x = r + a*kprev in LDS (bf16 for the WMMA B operand)
  for (int i = tid; i < N_TOT; i += 256) {
    xb[i] = __float2bfloat16(fmaf(a, kprev[i], r[i]));
  }
  __syncthreads();

  const int rowbase = blockIdx.x * 16;
  const int m  = lane & 15;          // row within the 16-row tile
  const int hi = (lane >> 4) & 1;    // lane group 0-15 vs 16-31

  // A layout (16-bit, 16x32): lanes 0-15 hold K {0..7,16..23}, lanes 16-31 hold K {8..15,24..31}
  const __hip_bfloat16* wrow = Wb + (size_t)(rowbase + m) * N_TOT + hi * 8;
  const int kbase = wave * (N_TOT / 8);

  v8f acc = {};
  #pragma unroll 4
  for (int kb = 0; kb < N_TOT / 8; kb += 32) {
    const int k0 = kbase + kb;
    TileU A, Bm;
    A.q[0] = *(const float4*)(wrow + k0);        // K +0..7   (or +8..15)
    A.q[1] = *(const float4*)(wrow + k0 + 16);   // K +16..23 (or +24..31)
    // B layout (16-bit, 32x16): lanes 0-15 hold K {0..7,16..23}, lanes 16-31 hold K {8..15,24..31};
    // every column replicates the x segment.
    const __hip_bfloat16* xp = xb + k0 + hi * 8;
    Bm.q[0] = *(const float4*)(xp);
    Bm.q[1] = *(const float4*)(xp + 16);
    acc = __builtin_amdgcn_wmma_f32_16x16x32_bf16(false, A.v, false, Bm.v,
                                                  (short)0, acc, false, false);
  }

  // Column N=0 of the 16x16 accumulator: lane 0 holds rows 0..7, lane 16 rows 8..15.
  if (lane == 0 || lane == 16) {
    float* dst = &red[wave][hi * 8];
    #pragma unroll
    for (int i = 0; i < 8; ++i) dst[i] = acc[i];
  }
  __syncthreads();

  if (tid < 16) {
    float s = 0.0f;
    #pragma unroll
    for (int w = 0; w < 8; ++w) s += red[w][tid];
    const int row = rowbase + tid;
    const float xv = fmaf(a, kprev[row], r[row]);   // fp32 recompute of x[row]
    float u = fmaxf(s + h[row], 0.0f);
    kout[row] = fmaf(u, u, -xv);                    // relu(.)^2 - x
  }
}

// ---- r += ds/6 * (k1 + 2k2 + 2k3 + k4); optionally emit output row ----
__global__ __launch_bounds__(256) void ssn_update(float* __restrict__ r,
                                                  const float* __restrict__ k1,
                                                  const float* __restrict__ k2,
                                                  const float* __restrict__ k3,
                                                  const float* __restrict__ k4,
                                                  const float* __restrict__ t,
                                                  int interval, int write_out,
                                                  float* __restrict__ out) {
  const int i = blockIdx.x * 256 + threadIdx.x;
  const float ds = (t[interval + 1] - t[interval]) * (1.0f / SUBSTEPS);
  const float rn = r[i] + (ds * (1.0f / 6.0f)) *
                   (k1[i] + 2.0f * k2[i] + 2.0f * k3[i] + k4[i]);
  r[i] = rn;
  if (write_out) out[(size_t)(interval + 1) * N_TOT + i] = rn;
}

extern "C" void kernel_launch(void* const* d_in, const int* in_sizes, int n_in,
                              void* d_out, int out_size, void* d_ws, size_t ws_size,
                              hipStream_t stream) {
  (void)in_sizes; (void)n_in; (void)out_size; (void)ws_size;
  const float* W  = (const float*)d_in[0];   // (N,N) fp32
  const float* h  = (const float*)d_in[1];   // (4,2048) fp32 -> flat N
  const float* r0 = (const float*)d_in[2];   // (4,2048) fp32 -> flat N
  const float* t  = (const float*)d_in[3];   // (L,) fp32
  float* out = (float*)d_out;                // (L, 4, 2048) fp32

  // Workspace layout: r, k1..k4 (5*N fp32), then 256B-aligned bf16 copy of W (128 MB).
  char* ws = (char*)d_ws;
  float* r  = (float*)ws;
  float* k1 = r  + N_TOT;
  float* k2 = k1 + N_TOT;
  float* k3 = k2 + N_TOT;
  float* k4 = k3 + N_TOT;
  size_t vec_bytes = ((size_t)5 * N_TOT * sizeof(float) + 255) & ~(size_t)255;
  __hip_bfloat16* Wb = (__hip_bfloat16*)(ws + vec_bytes);

  ssn_convert_w<<<8192, 256, 0, stream>>>(W, Wb, (long long)N_TOT * N_TOT);
  ssn_init<<<N_TOT / 256, 256, 0, stream>>>(r0, r, out);

  for (int iv = 0; iv < L_TS - 1; ++iv) {
    for (int s = 0; s < SUBSTEPS; ++s) {
      ssn_stage<<<N_TOT / 16, 256, 0, stream>>>(Wb, r, r,  h, t, iv, 0.0f, k1);
      ssn_stage<<<N_TOT / 16, 256, 0, stream>>>(Wb, r, k1, h, t, iv, 0.5f, k2);
      ssn_stage<<<N_TOT / 16, 256, 0, stream>>>(Wb, r, k2, h, t, iv, 0.5f, k3);
      ssn_stage<<<N_TOT / 16, 256, 0, stream>>>(Wb, r, k3, h, t, iv, 1.0f, k4);
      ssn_update<<<N_TOT / 256, 256, 0, stream>>>(r, k1, k2, k3, k4, t, iv,
                                                  s == SUBSTEPS - 1, out);
    }
  }
}
